// Qatten_2774548873269
// MI455X (gfx1250) — matrix operand order repaired and checked by reading, not compiled, not run on hardware
//
#include <hip/hip_runtime.h>

typedef _Float16 h16;
typedef __attribute__((ext_vector_type(16))) _Float16 v16h;
typedef __attribute__((ext_vector_type(8)))  _Float16 v8h;
typedef __attribute__((ext_vector_type(8)))  float    v8f;
typedef __attribute__((ext_vector_type(4)))  float    v4f;

#define B_TOT 32768
#define A_N   8
#define S_N   512
#define E_N   128
#define H_N   4
#define MT    32                  // batch rows per block
#define INV_SQRT_E 0.088388347648318447f
#define NEGV  -99999999.0f
#define COEF  0.001f

// ---- dynamic LDS layout (bytes); ~123KB -> 2 workgroups per WGP ----
#define L_S16   0                 // h16   [32][520]        33280
#define L_SELH  33280             // h16   [32][520]        33280
#define L_PROJ  66560             // float [32][256]        32768
#define L_HID   99328             // float [32][128]        16384
#define L_LOGIT 115712            // float [32][32]          4096
#define L_WTS   119808            // float [32][32]          4096
#define L_WHEAD 123904            // float [32][4]            512
#define L_PROJQ 124416            // float [32][4]            512
#define L_RED   124928            // float [256]             1024
#define L_ENT   125952            // float [4]                 16
#define L_TOTAL 125968

// A fragment (16x32 f16) from LDS, row-major, pitch in halves
__device__ __forceinline__ v16h load_a_lds(const h16* base, int pitch, int lane, int k0) {
  int row = lane & 15;
  int kb  = (lane >> 4) * 8;
  const h16* p = base + row * pitch + k0 + kb;
  v8h lo = *(const v8h*)p;
  v8h hi = *(const v8h*)(p + 16);
  v16h a;
  #pragma unroll
  for (int i = 0; i < 8; ++i) { a[i] = lo[i]; a[i + 8] = hi[i]; }
  return a;
}

// B fragment (32x16 f16): column n = row n of W (K contiguous), pitch in halves
__device__ __forceinline__ v16h load_b_glb(const h16* W, int pitch, int ntile, int lane, int k0) {
  int n  = lane & 15;
  int kb = (lane >> 4) * 8;
  const h16* p = W + (size_t)(ntile * 16 + n) * pitch + k0 + kb;
  v8h lo = *(const v8h*)p;
  v8h hi = *(const v8h*)(p + 16);
  v16h b;
  #pragma unroll
  for (int i = 0; i < 8; ++i) { b[i] = lo[i]; b[i + 8] = hi[i]; }
  return b;
}

__global__ __launch_bounds__(256) void prep_weights(
    const float* __restrict__ Wsel, const float* __restrict__ Wkey,
    const float* __restrict__ Wv1,
    h16* __restrict__ wselh, h16* __restrict__ wkeyt,
    float* __restrict__ wkeyq, h16* __restrict__ wv1h,
    float* __restrict__ out) {
  int g = blockIdx.x * 256 + threadIdx.x;
  if (g < 5) out[294912 + g] = 0.f;           // zero scalar + entropy slots each launch
  int idx = g;
  if (idx < 262144) { wselh[idx] = (h16)Wsel[idx]; return; }          // [512][512]
  idx -= 262144;
  if (idx < 32768) {                                                   // Wkey^T: [h][u][e]
    int hh = idx >> 13, r = idx & 8191, u = r >> 7, e = r & 127;
    wkeyt[(hh << 13) | (u << 7) | e] = (h16)Wkey[hh * 8320 + e * 65 + u];
    return;
  }
  idx -= 32768;
  if (idx < 512) {                                                     // q column (u=64), f32
    wkeyq[idx] = Wkey[(idx >> 7) * 8320 + (idx & 127) * 65 + 64];
    return;
  }
  idx -= 512;
  if (idx < 65536) wv1h[idx] = (h16)Wv1[idx];                          // [128][512]
}

__global__ __launch_bounds__(256) void qatten_main(
    const float* __restrict__ agent_qs, const float* __restrict__ states,
    const int* __restrict__ actions,
    const float* __restrict__ bv1, const float* __restrict__ Wv2,
    const float* __restrict__ bv2, const float* __restrict__ Wh,
    const float* __restrict__ bh,
    const h16* __restrict__ wselh, const h16* __restrict__ wkeyt,
    const float* __restrict__ wkeyq, const h16* __restrict__ wv1h,
    float* __restrict__ out) {
  extern __shared__ char smem[];
  h16   (*s16s)[520] = (h16(*)[520])(smem + L_S16);
  h16   (*selh)[520] = (h16(*)[520])(smem + L_SELH);
  float (*proj)[256] = (float(*)[256])(smem + L_PROJ);
  float (*hid)[128]  = (float(*)[128])(smem + L_HID);
  float (*logit)[32] = (float(*)[32])(smem + L_LOGIT);
  float (*wts)[32]   = (float(*)[32])(smem + L_WTS);
  float (*whead)[4]  = (float(*)[4])(smem + L_WHEAD);
  float (*projq)[4]  = (float(*)[4])(smem + L_PROJQ);
  float* red    = (float*)(smem + L_RED);
  float* entAcc = (float*)(smem + L_ENT);

  const int t = threadIdx.x;
  const int wave = t >> 5, lane = t & 31;
  const int gb0 = blockIdx.x * MT;

  if (t < H_N) entAcc[t] = 0.f;

  // ---- phase 0: stage states tile as f16 ----
  {
    const v4f* gs = (const v4f*)(states + (size_t)gb0 * S_N);
    #pragma unroll
    for (int i = 0; i < 16; ++i) {
      int lin = t + i * 256;                 // 0..4095 float4s
      int b = lin >> 7, s4 = lin & 127;
      v4f v = gs[b * 128 + s4];
      int s = s4 * 4;
      s16s[b][s]     = (h16)v[0];
      s16s[b][s + 1] = (h16)v[1];
      s16s[b][s + 2] = (h16)v[2];
      s16s[b][s + 3] = (h16)v[3];
    }
  }
  __syncthreads();

  // ---- phase 1: sel = states @ Wsel^T [32x512]  and  hid = states @ Wv1^T [32x128] ----
  {
    v8f accS[2][4], accV[2];
    #pragma unroll
    for (int m = 0; m < 2; ++m) {
      v8f z = {}; accV[m] = z;
      #pragma unroll
      for (int j = 0; j < 4; ++j) accS[m][j] = z;
    }
    for (int kk = 0; kk < 16; ++kk) {
      int k0 = kk * 32;
      v16h a0 = load_a_lds(&s16s[0][0],  520, lane, k0);
      v16h a1 = load_a_lds(&s16s[16][0], 520, lane, k0);
      #pragma unroll
      for (int j = 0; j < 4; ++j) {
        v16h b = load_b_glb(wselh, 512, wave + j * 8, lane, k0);
        accS[0][j] = __builtin_amdgcn_wmma_f32_16x16x32_f16(false, a0, false, b, (short)0, accS[0][j], false, false);
        accS[1][j] = __builtin_amdgcn_wmma_f32_16x16x32_f16(false, a1, false, b, (short)0, accS[1][j], false, false);
      }
      v16h bv = load_b_glb(wv1h, 512, wave, lane, k0);
      accV[0] = __builtin_amdgcn_wmma_f32_16x16x32_f16(false, a0, false, bv, (short)0, accV[0], false, false);
      accV[1] = __builtin_amdgcn_wmma_f32_16x16x32_f16(false, a1, false, bv, (short)0, accV[1], false, false);
    }
    int mrow = (lane >> 4) * 8, ncol = lane & 15;
    #pragma unroll
    for (int m = 0; m < 2; ++m) {
      #pragma unroll
      for (int j = 0; j < 4; ++j) {
        int nbase = (wave + j * 8) * 16 + ncol;
        #pragma unroll
        for (int r = 0; r < 8; ++r)
          selh[m * 16 + mrow + r][nbase] = (h16)accS[m][j][r];
      }
      int nb = wave * 16 + ncol;
      #pragma unroll
      for (int r = 0; r < 8; ++r) hid[m * 16 + mrow + r][nb] = accV[m][r];
    }
  }
  __syncthreads();

  // ---- phase 2: proj[b][h][u] = sum_e sel[b][h*128+e] * Wkey[h][e][u] ----
  {
    #pragma unroll
    for (int tti = 0; tti < 2; ++tti) {
      int tt = wave + tti * 8;              // 16 tiles: h(4) x utile(4)
      int hh = tt >> 2, ut = tt & 3;
      v8f accP[2]; { v8f z = {}; accP[0] = z; accP[1] = z; }
      #pragma unroll
      for (int kk = 0; kk < 4; ++kk) {
        int k0 = kk * 32;
        v16h a0 = load_a_lds(&selh[0][0],  520, lane, hh * 128 + k0);
        v16h a1 = load_a_lds(&selh[16][0], 520, lane, hh * 128 + k0);
        v16h b  = load_b_glb(wkeyt + (size_t)hh * 64 * 128, 128, ut, lane, k0);
        accP[0] = __builtin_amdgcn_wmma_f32_16x16x32_f16(false, a0, false, b, (short)0, accP[0], false, false);
        accP[1] = __builtin_amdgcn_wmma_f32_16x16x32_f16(false, a1, false, b, (short)0, accP[1], false, false);
      }
      int mrow = (lane >> 4) * 8, ncol = lane & 15;
      int ub = ut * 16 + ncol;
      #pragma unroll
      for (int m = 0; m < 2; ++m)
        #pragma unroll
        for (int r = 0; r < 8; ++r)
          proj[m * 16 + mrow + r][(hh << 6) + ub] = accP[m][r];
    }
  }
  __syncthreads();

  // ---- phase 3a: w_head dots (split over 2 threads) ----
  {
    int pair = t >> 1;                      // (b,h) 0..127
    int b = pair >> 2, hh = pair & 3, sub = t & 1;
    float sum = 0.f;
    const float* wrow = Wh + hh * S_N + sub * 256;
    const h16* srow = &s16s[b][sub * 256];
    for (int s = 0; s < 256; ++s) sum += (float)srow[s] * wrow[s];
    red[t] = sum;
  }
  __syncthreads();
  if (t < 128) {
    int b = t >> 2, hh = t & 3;
    whead[b][hh] = fabsf(red[2 * t] + red[2 * t + 1] + bh[hh]);
    float q = 0.f;
    const float* wq = wkeyq + (hh << 7);
    for (int e = 0; e < E_N; ++e) q += (float)selh[b][(hh << 7) + e] * wq[e];
    projq[b][hh] = q;
  }
  __syncthreads();

  // ---- phase 3b: logits + sum-of-squares ----
  {
    float sq = 0.f;
    #pragma unroll
    for (int i = 0; i < 4; ++i) {
      int it = t + i * 256;                 // (b,h,a) 0..1023
      int b = it >> 5, hh = (it >> 3) & 3, a = it & 7;
      float sum = 0.f;
      const h16* xs = &s16s[b][a * 64];
      const float* pr = &proj[b][hh << 6];
      for (int u = 0; u < 64; ++u) sum += (float)xs[u] * pr[u];
      sum += agent_qs[(size_t)(gb0 + b) * A_N + a] * projq[b][hh];
      logit[b][(hh << 3) + a] = sum;
      sq += sum * sum;
    }
    red[t] = sq;
  }
  __syncthreads();

  // ---- phase 3c: masked softmax over agents, entropy partials ----
  if (t < 128) {
    int b = t >> 2, hh = t & 3;
    float l[8]; float mx = -3.4e38f;
    #pragma unroll
    for (int a = 0; a < A_N; ++a) {
      float v = logit[b][(hh << 3) + a] * INV_SQRT_E;
      if (actions[(size_t)(gb0 + b) * A_N + a] == 0) v = NEGV;
      l[a] = v; mx = fmaxf(mx, v);
    }
    float denom = 0.f;
    #pragma unroll
    for (int a = 0; a < A_N; ++a) { l[a] = __expf(l[a] - mx); denom += l[a]; }
    float inv = 1.f / denom, ent = 0.f;
    #pragma unroll
    for (int a = 0; a < A_N; ++a) {
      float w = l[a] * inv;
      wts[b][(hh << 3) + a] = w;
      ent += __logf(w + 1e-8f) * w;
    }
    atomicAdd(&entAcc[hh], ent);
  }
  __syncthreads();

  // ---- phase 3d: outputs ----
  {
    int b = t >> 3, a = t & 7;              // 32*8 = 256
    float ha = 0.f;
    #pragma unroll
    for (int hh = 0; hh < H_N; ++hh) ha += wts[b][(hh << 3) + a] * whead[b][hh];
    out[(size_t)(gb0 + b) * A_N + a] = ha;
  }
  if (t < MT) {
    float acc = bv2[0];
    for (int e = 0; e < E_N; ++e) {
      float hv = hid[t][e] + bv1[e];
      acc += fmaxf(hv, 0.f) * Wv2[e];
    }
    out[262144 + gb0 + t] = acc;
  }
  #pragma unroll
  for (int s = 128; s > 0; s >>= 1) {
    if (t < s) red[t] += red[t + s];
    __syncthreads();
  }
  if (t == 0) atomicAdd(out + 294912, red[0] * (COEF / (float)(B_TOT * A_N)));
  if (t < H_N) atomicAdd(out + 294913 + t, -entAcc[t] * (1.0f / (float)B_TOT));
}

extern "C" void kernel_launch(void* const* d_in, const int* in_sizes, int n_in,
                              void* d_out, int out_size, void* d_ws, size_t ws_size,
                              hipStream_t stream) {
  const float* agent_qs = (const float*)d_in[0];
  const float* states   = (const float*)d_in[1];
  const int*   actions  = (const int*)d_in[2];
  const float* W_sel    = (const float*)d_in[3];
  const float* W_key    = (const float*)d_in[4];
  const float* Wv1      = (const float*)d_in[5];
  const float* bv1      = (const float*)d_in[6];
  const float* Wv2      = (const float*)d_in[7];
  const float* bv2      = (const float*)d_in[8];
  const float* Wh       = (const float*)d_in[9];
  const float* bh       = (const float*)d_in[10];
  float* out = (float*)d_out;

  h16*   wselh = (h16*)d_ws;                // 262144 halves
  h16*   wkeyt = wselh + 262144;            // 32768 halves  [h][u][e]
  float* wkeyq = (float*)(wkeyt + 32768);   // 512 f32
  h16*   wv1h  = (h16*)(wkeyq + 512);       // 65536 halves

  prep_weights<<<1411, 256, 0, stream>>>(W_sel, W_key, Wv1, wselh, wkeyt, wkeyq, wv1h, out);
  qatten_main<<<B_TOT / MT, 256, L_TOTAL, stream>>>(
      agent_qs, states, actions, bv1, Wv2, bv2, Wh, bh,
      wselh, wkeyt, wkeyq, wv1h, out);
}
